// Attention_10874857193754
// MI455X (gfx1250) — matrix operand order repaired and checked by reading, not compiled, not run on hardware
//
#include <hip/hip_runtime.h>
#include <hip/hip_bf16.h>

// ---------------------------------------------------------------------------
// Bahdanau additive attention, fused for MI455X (gfx1250, wave32, WMMA).
// B=256, L=1024, D=H=U=512.
//   k0: W1 [D,U] fp32 -> W1t [U,D] bf16 (workspace, L2-resident, 512 KB)
//   k1: proj_h = hidden@W2 + b2        (staged in d_out context region)
//   k2: scores[b,l] = tanh(features@W1 + proj_h + b1) . V_w   (bf16 WMMA)
//       64-row M-blocking: 4x4 accumulator tiles per wave; B fragments
//       reused across 4 M-tiles (4x less L2 traffic on W1t).
//       Hardware v_tanh_f32 when available; float4->2xbf16 packed LDS fill.
//   k3: softmax over L (in place, d_out weights region)
//   k4: context[b,d] = sum_l w[b,l] * features[b,l,d]
// ---------------------------------------------------------------------------

typedef __attribute__((ext_vector_type(16))) __bf16 v16bf;
typedef __attribute__((ext_vector_type(8)))  __bf16 v8bf;
typedef __attribute__((ext_vector_type(8)))  float  v8f;

#define B_  256
#define L_  1024
#define D_  512
#define U_  512
#define LP  520   // LDS row stride in halves: 520*2=1040B = 65*16B -> aligned, conflict-free
#define MT  4     // M-tiles (16 rows each) per block = 64 rows

#if __has_builtin(__builtin_amdgcn_tanhf)
#define TANHF(x) __builtin_amdgcn_tanhf(x)   // hardware v_tanh_f32 (TRANS unit)
#else
#define TANHF(x) tanhf(x)
#endif

__device__ __forceinline__ __bf16 f32_to_bf16(float f) {
  unsigned u = __builtin_bit_cast(unsigned, f);
  u += 0x7FFFu + ((u >> 16) & 1u);              // round-to-nearest-even
  unsigned short h = (unsigned short)(u >> 16);
  return __builtin_bit_cast(__bf16, h);
}

__device__ __forceinline__ unsigned pack_bf16x2(float a, float b) {
  unsigned ua = __builtin_bit_cast(unsigned, a);
  unsigned ub = __builtin_bit_cast(unsigned, b);
  ua += 0x7FFFu + ((ua >> 16) & 1u);            // RNE per element
  ub += 0x7FFFu + ((ub >> 16) & 1u);
  return (ua >> 16) | (ub & 0xFFFF0000u);
}

// ---------------- k0: transpose + convert W1 -> bf16 [U][D] ----------------
__global__ __launch_bounds__(256) void w1t_kernel(const float* __restrict__ W1,
                                                  __bf16* __restrict__ W1t) {
  int idx = blockIdx.x * 256 + threadIdx.x;     // 512*512 elements, coalesced read
  int d = idx >> 9;
  int u = idx & 511;
  W1t[u * D_ + d] = f32_to_bf16(W1[idx]);       // W1 row-major [D][U]
}

// ---------------- k1: proj_h = hidden @ W2 + W2_b --------------------------
__global__ __launch_bounds__(256) void projh_kernel(const float* __restrict__ hidden,
                                                    const float* __restrict__ W2,
                                                    const float* __restrict__ W2b,
                                                    float* __restrict__ proj_h) {
  __shared__ float h_s[512];
  int b = blockIdx.x;
  int tid = threadIdx.x;
  for (int i = tid; i < 512; i += 256) h_s[i] = hidden[b * 512 + i];
  __syncthreads();
  for (int u = tid; u < U_; u += 256) {
    float acc = W2b[u];
    #pragma unroll 4
    for (int k = 0; k < 512; ++k) acc = fmaf(h_s[k], W2[k * U_ + u], acc);
    proj_h[b * U_ + u] = acc;
  }
}

// ---------------- k2: fused features@W1 -> tanh -> .V  (bf16 WMMA) ---------
// grid = B * (L/64) blocks, 256 threads = 8 waves. Wave w owns U cols [64w,64w+64).
// Each block computes a 64(M) x 512(U) proj_f tile entirely in registers.
__global__ __launch_bounds__(256) void score_kernel(const float*  __restrict__ features,
                                                    const __bf16* __restrict__ W1t,
                                                    const float*  __restrict__ W1b,
                                                    const float*  __restrict__ proj_h,
                                                    const float*  __restrict__ Vw,
                                                    const float*  __restrict__ Vb,
                                                    float* __restrict__ scores) {
  __shared__ __bf16 As[MT * 16 * LP];   // 64 feature rows x 512 K, bf16, padded
  __shared__ float  s_part[8][MT * 16]; // per-wave partial row scores

  int tid  = threadIdx.x;
  int wave = tid >> 5;
  int lane = tid & 31;
  int blk  = blockIdx.x;
  int b    = blk >> 4;                  // L/64 = 16 tiles per batch
  int lt   = blk & 15;

  // Stage 1: cooperative fp32 -> bf16 tile fill, float4 loads + uint2 stores.
  const float4* frow4 =
      (const float4*)(features + ((size_t)b * L_ + (size_t)lt * 64) * D_);
  for (int i = tid; i < 64 * (D_ / 4); i += 256) {
    int r = i >> 7, c4 = i & 127;       // 128 float4 per row
    float4 v = frow4[i];
    uint2 pk;
    pk.x = pack_bf16x2(v.x, v.y);
    pk.y = pack_bf16x2(v.z, v.w);
    *(uint2*)&As[r * LP + c4 * 4] = pk; // 8B-aligned (1040*r + 8*c4)
  }
  __syncthreads();

  int m  = lane & 15;                   // A row / B column within tile
  int kh = lane >> 4;                   // K half selector: k = 16*kh + s (contiguous, A and B alike)
  int u0 = wave * 64;

  v8f zero = {0.f, 0.f, 0.f, 0.f, 0.f, 0.f, 0.f, 0.f};
  v8f acc[4][MT];
  #pragma unroll
  for (int ut = 0; ut < 4; ++ut)
    #pragma unroll
    for (int mt = 0; mt < MT; ++mt) acc[ut][mt] = zero;

  for (int kb = 0; kb < D_; kb += 32) {
    // A fragments (16x32 bf16 each): lane holds K block kh*16..+15, contiguous
    v16bf af[MT];
    #pragma unroll
    for (int mt = 0; mt < MT; ++mt) {
      const __bf16* arow = &As[(mt * 16 + m) * LP + kb];
      v8bf a0 = *(const v8bf*)(arow + kh * 16);
      v8bf a1 = *(const v8bf*)(arow + kh * 16 + 8);
      af[mt] = __builtin_shufflevector(a0, a1,
          0,1,2,3,4,5,6,7,8,9,10,11,12,13,14,15);
    }
    #pragma unroll
    for (int ut = 0; ut < 4; ++ut) {
      // B fragment (32x16): lane holds col n, K block = kh*16..+15, contiguous in W1t
      int n = u0 + ut * 16 + m;
      v16bf bfrag = *(const v16bf*)(W1t + (size_t)n * D_ + kb + kh * 16);
      #pragma unroll
      for (int mt = 0; mt < MT; ++mt)
        acc[ut][mt] = __builtin_amdgcn_wmma_f32_16x16x32_bf16(
            false, af[mt], false, bfrag, (short)0, acc[ut][mt], false, false);
    }
  }

  // Epilogue: tanh(acc + bias) * Vw, reduce over U.
  // C layout: VGPR r, lane -> row M = mt*16 + r + 8*(lane>>4), col u = u0+ut*16+(lane&15)
  float partial[MT][8];
  #pragma unroll
  for (int mt = 0; mt < MT; ++mt)
    #pragma unroll
    for (int r = 0; r < 8; ++r) partial[mt][r] = 0.f;

  #pragma unroll
  for (int ut = 0; ut < 4; ++ut) {
    int u = u0 + ut * 16 + m;
    float bias = W1b[u] + proj_h[(size_t)b * U_ + u];
    float vw = Vw[u];
    #pragma unroll
    for (int mt = 0; mt < MT; ++mt)
      #pragma unroll
      for (int r = 0; r < 8; ++r)
        partial[mt][r] += TANHF(acc[ut][mt][r] + bias) * vw;
  }
  // Sum across the 16 lanes of each half-wave (distinct u's, same row group).
  #pragma unroll
  for (int mt = 0; mt < MT; ++mt) {
    #pragma unroll
    for (int r = 0; r < 8; ++r) {
      float v = partial[mt][r];
      v += __shfl_xor(v, 1, 32);
      v += __shfl_xor(v, 2, 32);
      v += __shfl_xor(v, 4, 32);
      v += __shfl_xor(v, 8, 32);
      partial[mt][r] = v;
    }
  }
  if ((lane & 15) == 0) {
    int rbase = (lane >> 4) * 8;
    #pragma unroll
    for (int mt = 0; mt < MT; ++mt)
      #pragma unroll
      for (int r = 0; r < 8; ++r)
        s_part[wave][mt * 16 + rbase + r] = partial[mt][r];
  }
  __syncthreads();
  if (tid < 64) {                       // deterministic cross-wave sum
    float s = Vb[0];
    #pragma unroll
    for (int w = 0; w < 8; ++w) s += s_part[w][tid];
    scores[(size_t)b * L_ + lt * 64 + tid] = s;
  }
}

// ---------------- k3: softmax over L, in place -----------------------------
__global__ __launch_bounds__(256) void softmax_kernel(float* __restrict__ sw) {
  __shared__ float red[8];
  __shared__ float s_max, s_inv;
  int b = blockIdx.x, tid = threadIdx.x, lane = tid & 31, wave = tid >> 5;
  float* p = sw + (size_t)b * L_;
  float v[4];
  float lmax = -3.4e38f;
  #pragma unroll
  for (int j = 0; j < 4; ++j) { v[j] = p[tid + 256 * j]; lmax = fmaxf(lmax, v[j]); }
  #pragma unroll
  for (int msk = 16; msk >= 1; msk >>= 1) lmax = fmaxf(lmax, __shfl_xor(lmax, msk, 32));
  if (lane == 0) red[wave] = lmax;
  __syncthreads();
  if (tid == 0) {
    float mx = red[0];
    for (int w = 1; w < 8; ++w) mx = fmaxf(mx, red[w]);
    s_max = mx;
  }
  __syncthreads();
  float mx = s_max, lsum = 0.f;
  #pragma unroll
  for (int j = 0; j < 4; ++j) { v[j] = __expf(v[j] - mx); lsum += v[j]; }
  #pragma unroll
  for (int msk = 16; msk >= 1; msk >>= 1) lsum += __shfl_xor(lsum, msk, 32);
  __syncthreads();                      // red[] reuse
  if (lane == 0) red[wave] = lsum;
  __syncthreads();
  if (tid == 0) {
    float s = 0.f;
    for (int w = 0; w < 8; ++w) s += red[w];
    s_inv = 1.0f / s;
  }
  __syncthreads();
  float inv = s_inv;
  #pragma unroll
  for (int j = 0; j < 4; ++j) p[tid + 256 * j] = v[j] * inv;
}

// ---------------- k4: context = sum_l w[b,l] * features[b,l,:] -------------
// One block per batch row; each lane owns 2 consecutive d's (float2 loads).
__global__ __launch_bounds__(256) void context_kernel(const float* __restrict__ features,
                                                      const float* __restrict__ weights,
                                                      float* __restrict__ ctx) {
  __shared__ float w_s[L_];
  int b = blockIdx.x;
  int tid = threadIdx.x;
  for (int i = tid; i < L_; i += 256) w_s[i] = weights[(size_t)b * L_ + i];
  __syncthreads();
  const float2* f = (const float2*)(features + (size_t)b * L_ * D_) + tid;
  float2 acc; acc.x = 0.f; acc.y = 0.f;
  #pragma unroll 4
  for (int l = 0; l < L_; ++l) {
    float2 v = f[(size_t)l * (D_ / 2)];
    acc.x = fmaf(w_s[l], v.x, acc.x);
    acc.y = fmaf(w_s[l], v.y, acc.y);
  }
  ((float2*)(ctx + (size_t)b * D_))[tid] = acc;
}

// ---------------------------------------------------------------------------
extern "C" void kernel_launch(void* const* d_in, const int* in_sizes, int n_in,
                              void* d_out, int out_size, void* d_ws, size_t ws_size,
                              hipStream_t stream) {
  const float* features = (const float*)d_in[0];   // [B,L,D]
  const float* hidden   = (const float*)d_in[1];   // [B,H]
  const float* W1_w     = (const float*)d_in[2];   // [D,U]
  const float* W1_b     = (const float*)d_in[3];   // [U]
  const float* W2_w     = (const float*)d_in[4];   // [H,U]
  const float* W2_b     = (const float*)d_in[5];   // [U]
  const float* V_w      = (const float*)d_in[6];   // [U,1]
  const float* V_b      = (const float*)d_in[7];   // [1]

  float* out = (float*)d_out;
  float* ctx = out;                 // [B,D] context (also proj_h staging)
  float* wts = out + B_ * D_;       // [B,L] scores -> weights (in place)
  __bf16* W1t = (__bf16*)d_ws;      // [U,D] bf16, 512 KB

  w1t_kernel   <<<(D_ * U_) / 256, 256, 0, stream>>>(W1_w, W1t);
  projh_kernel <<<B_, 256, 0, stream>>>(hidden, W2_w, W2_b, ctx);
  score_kernel <<<B_ * (L_ / 64), 256, 0, stream>>>(features, W1t, W1_b, ctx, V_w, V_b, wts);
  softmax_kernel<<<B_, 256, 0, stream>>>(wts);
  context_kernel<<<B_, 256, 0, stream>>>(features, wts, ctx);
}
